// InformationNode_35966056136967
// MI455X (gfx1250) — matrix operand order repaired and checked by reading, not compiled, not run on hardware
//
#include <hip/hip_runtime.h>
#include <hip/hip_bf16.h>

typedef __attribute__((ext_vector_type(16))) __bf16 v16bf;
typedef __attribute__((ext_vector_type(8)))  __bf16 v8bf;
typedef __attribute__((ext_vector_type(8)))  float  v8f;
typedef __attribute__((ext_vector_type(4)))  float  v4f;
typedef __attribute__((ext_vector_type(4)))  int    v4i;

constexpr int kBatch = 32768;
constexpr int kNbr   = 8;
constexpr int kDin   = 256;
constexpr int kDh    = 512;
constexpr int kDout  = 256;
constexpr int kBT    = 16;            // batch rows per workgroup = one WMMA M-tile
constexpr int kNT1   = kDh  / 16;     // 32 GEMM1 N-tiles
constexpr int kK1    = kDin / 32;     // 8  GEMM1 K-steps
constexpr int kNT2   = kDout / 16;    // 16 GEMM2 N-tiles
constexpr int kK2    = kDh  / 32;     // 16 GEMM2 K-steps
constexpr int kWaves = 1 + kNbr;      // wave 0 = local state, waves 1..8 = neighbors
constexpr int kThreads = kWaves * 32; // 288 threads, wave32

// Dynamic LDS pool layout (bytes):
//   [0, 65536)          : two 32 KB weight slices (double buffer)  -- loop phase
//   [0, 16384)          : float acc[16*256]  (REUSED after the loop)
//   [65536, 74752)      : bf16 stage[9][512] (h-chunk transpose)
//   [74752, 75264)      : float sc[8][16]    (attention scores)
constexpr int kSliceBf  = 16384;                  // bf16 elems per k2 slice (32 KB)
constexpr int kStageOff = 2 * kSliceBf * 2;       // 65536 B
constexpr int kScOff    = kStageOff + kWaves * 512 * 2;  // 74752 B
constexpr int kDynBytes = kScOff + kNbr * kBT * 4;       // 75264 B

// ---------------------------------------------------------------------------
// CDNA5 async global->LDS copy (ASYNCcnt-tracked), with a safe fallback.
// Probe-learned signature: (global int4*, local int4*, imm offset, imm cpol).
// ---------------------------------------------------------------------------
#define K_HAS_ASYNC_LDS __has_builtin(__builtin_amdgcn_global_load_async_to_lds_b128)

__device__ __forceinline__ void async_cp16(const __bf16* g, __bf16* l) {
#if K_HAS_ASYNC_LDS
  typedef __attribute__((address_space(1))) v4i gv4i;
  typedef __attribute__((address_space(3))) v4i lv4i;
  __builtin_amdgcn_global_load_async_to_lds_b128((gv4i*)g, (lv4i*)l, 0, 0);
#else
  *(v8bf*)l = *(const v8bf*)g;           // synchronous fallback via VGPRs
#endif
}

template <int N>
__device__ __forceinline__ void wait_asynccnt() {
#if __has_builtin(__builtin_amdgcn_s_wait_asynccnt)
  __builtin_amdgcn_s_wait_asynccnt(N);
#else
  asm volatile("s_wait_asynccnt %0" ::"i"(N) : "memory");
#endif
}

// ---------------------------------------------------------------------------
// Pack a row-major f32 weight matrix W[K][N] into bf16 WMMA B-fragment order:
//   Wf[((kk*NT + nt)*32 + lane)*16 + e]  where K = kk*32 + 8*(lane/16) + e for e < 8,
//   K = kk*32 + 16 + 8*(lane/16) + (e - 8) for e >= 8, and N = nt*16 + (lane & 15).
// ---------------------------------------------------------------------------
__global__ __launch_bounds__(256) void pack_wmma_b(const float* __restrict__ W,
                                                   unsigned short* __restrict__ Wf,
                                                   int K, int Nn) {
  int tid = blockIdx.x * blockDim.x + threadIdx.x;
  int total = K * Nn;
  if (tid >= total) return;
  int NT   = Nn >> 4;
  int e    = tid & 15;
  int lane = (tid >> 4) & 31;
  int frag = tid >> 9;
  int nt   = frag % NT;
  int kk   = frag / NT;
  int k = kk * 32 + (e & 8) * 2 + ((lane & 16) ? 8 : 0) + (e & 7);
  int n = nt * 16 + (lane & 15);
  __bf16 v = (__bf16)W[k * Nn + n];
  ((__bf16*)Wf)[tid] = v;
}

// ---------------------------------------------------------------------------
// Fully fused: LayerNorm -> Linear(256,512)+ReLU -> Linear(512,256) for the
// local row and all 8 neighbor rows of a 16-wide batch tile, then softmax
// attention over neighbors and aggregation, writing only the final output.
// Per-k2 weight slices are async-DMAed into a double-buffered LDS pool
// (one barrier per iteration; loads fly under the previous compute phase).
// ---------------------------------------------------------------------------
__global__ __launch_bounds__(kThreads) void node_fused(
    const float* __restrict__ state,       // [B, 256]
    const float* __restrict__ neigh,       // [8, B, 256]
    const float* __restrict__ gamma,       // [256]
    const float* __restrict__ beta,        // [256]
    const float* __restrict__ b1,          // [512]
    const float* __restrict__ b2,          // [256]
    const float* __restrict__ attn_w,      // [256]
    const float* __restrict__ attn_b,      // [1]
    const unsigned short* __restrict__ W1f,// bf16 packed [8*32 frags][32][16]
    const unsigned short* __restrict__ W2f,// bf16 packed [16*16 frags][32][16]
    float* __restrict__ out)               // [B, 256]
{
  extern __shared__ __align__(16) char smem[];
  __bf16* wpool = (__bf16*)smem;                 // 2 x 16384 bf16 slices
  __bf16* stage = (__bf16*)(smem + kStageOff);   // 9 x 512 bf16
  float*  sc    = (float*)(smem + kScOff);       // 8 x 16
  float*  acc   = (float*)smem;                  // overlays wpool, post-loop

  const int tid   = threadIdx.x;
  const int wave  = tid >> 5;          // 0..8 (source index)
  const int lane  = tid & 31;
  const int lhalf = lane >> 4;         // 0: rows/K-low half, 1: high half
  const int lm    = lane & 15;
  const int b0    = blockIdx.x * kBT;

  const __bf16* w1 = reinterpret_cast<const __bf16*>(W1f);
  const __bf16* w2 = reinterpret_cast<const __bf16*>(W2f);

  // ---- Load this wave's 16x256 input tile in A-fragment element order ------
  const float* src = (wave == 0)
      ? (state + (size_t)(b0 + lm) * kDin)
      : (neigh + ((size_t)(wave - 1) * kBatch + b0 + lm) * kDin);

  v16bf A1[kK1];
  float sum = 0.f, ssq = 0.f;
#pragma unroll
  for (int k = 0; k < kK1; ++k) {
    const float* cp = src + k * 32 + lhalf * 8;
#pragma unroll
    for (int c = 0; c < 2; ++c) {          // two contiguous 8-float chunks
      v4f a = *(const v4f*)(cp + c * 16);
      v4f b = *(const v4f*)(cp + c * 16 + 4);
#pragma unroll
      for (int j = 0; j < 4; ++j) {
        float v0 = a[j], v1 = b[j];
        sum += v0 + v1;
        ssq += v0 * v0 + v1 * v1;
        A1[k][c * 8 + j]     = (__bf16)v0;
        A1[k][c * 8 + 4 + j] = (__bf16)v1;
      }
    }
  }
  sum += __shfl_xor(sum, 16, 32);
  ssq += __shfl_xor(ssq, 16, 32);
  const float mu  = sum * (1.f / kDin);
  const float var = ssq * (1.f / kDin) - mu * mu;
  const float rs  = rsqrtf(var + 1e-5f);

  // In-place normalize + affine on the bf16 fragments.
#pragma unroll
  for (int k = 0; k < kK1; ++k) {
    const float* gp = gamma + k * 32 + lhalf * 8;
    const float* bp = beta  + k * 32 + lhalf * 8;
#pragma unroll
    for (int c = 0; c < 2; ++c) {
#pragma unroll
      for (int j = 0; j < 8; ++j) {
        int e = c * 8 + j;
        float xv = (float)A1[k][e];
        A1[k][e] = (__bf16)((xv - mu) * rs * gp[c * 16 + j] + bp[c * 16 + j]);
      }
    }
  }

  // ---- Async weight staging: 2048 x 16B chunks, threads 0..255 x 8 chunks --
  // Chunk c < 1024 -> B1 slice half, else B2 slice half of iteration k2v.
  auto stage_slice = [&](int k2v, __bf16* dst) {
    if (tid < 256) {
#pragma unroll
      for (int i = 0; i < 8; ++i) {
        int c = tid + i * 256;
        const __bf16* g;
        __bf16* l;
        if (i < 4) {
          int f = c >> 6, w = c & 63;          // f: 0..15 => (sub, k1)
          int k1 = f & 7, sub = f >> 3;
          g = w1 + ((size_t)(k1 * kNT1 + 2 * k2v + sub)) * 512 + w * 8;
          l = dst + f * 512 + w * 8;
        } else {
          int c2 = c - 1024;
          int f = c2 >> 6, w = c2 & 63;        // f = nt2
          g = w2 + ((size_t)(k2v * kNT2 + f)) * 512 + w * 8;
          l = dst + 8192 + f * 512 + w * 8;
        }
        async_cp16(g, l);
      }
    }
  };

  // ---- Fused GEMM1 (LN(x) @ W1, ReLU) -> GEMM2 (h @ W2) --------------------
  v8f C2[kNT2];
  {
    v8f z = {};
#pragma unroll
    for (int t = 0; t < kNT2; ++t) C2[t] = z;
  }
  __bf16* st = stage + wave * 512;

  stage_slice(0, wpool);                       // prime the pipeline

#pragma unroll 1
  for (int k2 = 0; k2 < kK2; ++k2) {
    wait_asynccnt<0>();        // this wave's slice-k2 DMAs have landed in LDS
    __syncthreads();           // all waves' DMAs visible; k2-1 consumers done
    if (k2 + 1 < kK2)          // next slice flies under this compute phase
      stage_slice(k2 + 1, wpool + ((k2 + 1) & 1) * kSliceBf);

    const __bf16* cur = wpool + (k2 & 1) * kSliceBf;
    const __bf16* wB1 = cur;
    const __bf16* wB2 = cur + 8192;

    // Two 16-wide GEMM1 tiles produce h[:, k2*32 .. k2*32+31].
#pragma unroll
    for (int sub = 0; sub < 2; ++sub) {
      const int nt1 = k2 * 2 + sub;
      v8f c1 = {};
#pragma unroll
      for (int k1 = 0; k1 < kK1; ++k1) {
        v16bf bfrag = *(const v16bf*)(wB1 + (sub * 8 + k1) * 512 + lane * 16);
        c1 = __builtin_amdgcn_wmma_f32_16x16x32_bf16(
                 false, A1[k1], false, bfrag, (short)0, c1, false, false);
      }
      const float bias1 = b1[nt1 * 16 + lm];
#pragma unroll
      for (int r = 0; r < 8; ++r) {
        float v = c1[r] + bias1;
        v = v > 0.f ? v : 0.f;
        int row = r + lhalf * 8;                 // C layout -> (row, col=lm)
        st[row * 32 + sub * 16 + lm] = (__bf16)v;
      }
    }
    // Re-read the 16x32 h chunk as a GEMM2 A fragment (wave-local LDS).
    v8bf lo = *(const v8bf*)(st + lm * 32 + lhalf * 8);
    v8bf hi = *(const v8bf*)(st + lm * 32 + lhalf * 8 + 16);
    v16bf A2;
#pragma unroll
    for (int j = 0; j < 8; ++j) { A2[j] = lo[j]; A2[8 + j] = hi[j]; }

#pragma unroll
    for (int nt2 = 0; nt2 < kNT2; ++nt2) {
      v16bf bfrag = *(const v16bf*)(wB2 + nt2 * 512 + lane * 16);
      C2[nt2] = __builtin_amdgcn_wmma_f32_16x16x32_bf16(
                    false, A2, false, bfrag, (short)0, C2[nt2], false, false);
    }
  }

  // ---- Bias2 + attention score partials (registers only) -------------------
  float p[8];
#pragma unroll
  for (int r = 0; r < 8; ++r) p[r] = 0.f;
#pragma unroll
  for (int nt2 = 0; nt2 < kNT2; ++nt2) {
    const float bias2 = b2[nt2 * 16 + lm];
    const float aw    = attn_w[nt2 * 16 + lm];
#pragma unroll
    for (int r = 0; r < 8; ++r) {
      float v = C2[nt2][r] + bias2;
      C2[nt2][r] = v;
      p[r] += v * aw;
    }
  }
  // Reduce over the 16 lanes of each half (rows lhalf*8 + r).
#pragma unroll
  for (int m = 1; m < 16; m <<= 1) {
#pragma unroll
    for (int r = 0; r < 8; ++r) p[r] += __shfl_xor(p[r], m, 32);
  }

  __syncthreads();             // weight pool now dead -> safe to overlay acc
  for (int i = tid; i < kBT * kDout; i += kThreads) acc[i] = 0.f;
  if (wave >= 1 && lm == 0) {
#pragma unroll
    for (int r = 0; r < 8; ++r) sc[(wave - 1) * kBT + lhalf * 8 + r] = p[r] + attn_b[0];
  }
  __syncthreads();

  // ---- Softmax over neighbors; wave 0 (local) uses weight 1 ----------------
  float wgt[8];
  if (wave == 0) {
#pragma unroll
    for (int r = 0; r < 8; ++r) wgt[r] = 1.f;
  } else {
#pragma unroll
    for (int r = 0; r < 8; ++r) {
      const int row = lhalf * 8 + r;
      float m = sc[row];
#pragma unroll
      for (int n = 1; n < kNbr; ++n) m = fmaxf(m, sc[n * kBT + row]);
      float Z = 0.f;
#pragma unroll
      for (int n = 0; n < kNbr; ++n) Z += __expf(sc[n * kBT + row] - m);
      wgt[r] = __expf(sc[(wave - 1) * kBT + row] - m) / Z;
    }
  }

  // ---- Weighted aggregation across the 9 waves via LDS float atomics -------
#pragma unroll
  for (int nt2 = 0; nt2 < kNT2; ++nt2) {
#pragma unroll
    for (int r = 0; r < 8; ++r) {
      atomicAdd(&acc[(lhalf * 8 + r) * kDout + nt2 * 16 + lm],
                wgt[r] * C2[nt2][r]);
    }
  }
  __syncthreads();

  // ---- Coalesced write of the finished 16x256 output tile ------------------
  for (int i = tid; i < kBT * kDout; i += kThreads) {
    int row = i >> 8, col = i & 255;
    out[(size_t)(b0 + row) * kDout + col] = acc[i];
  }
}

// ---------------------------------------------------------------------------
extern "C" void kernel_launch(void* const* d_in, const int* in_sizes, int n_in,
                              void* d_out, int out_size, void* d_ws, size_t ws_size,
                              hipStream_t stream) {
  const float* state = (const float*)d_in[0];
  const float* neigh = (const float*)d_in[1];
  const float* gamma = (const float*)d_in[2];
  const float* beta  = (const float*)d_in[3];
  const float* W1    = (const float*)d_in[4];
  const float* b1    = (const float*)d_in[5];
  const float* W2    = (const float*)d_in[6];
  const float* b2    = (const float*)d_in[7];
  const float* aw    = (const float*)d_in[8];
  const float* ab    = (const float*)d_in[9];
  float* out = (float*)d_out;

  unsigned short* W1f = (unsigned short*)d_ws;                // 256 KB bf16
  unsigned short* W2f = W1f + (size_t)kDin * kDh;             // 256 KB bf16

  // gfx1250 WGPs have 320 KB LDS; raise the dynamic-LDS cap for our 73.5 KB.
  (void)hipFuncSetAttribute((const void*)node_fused,
                            hipFuncAttributeMaxDynamicSharedMemorySize,
                            kDynBytes);

  {
    int total = kDin * kDh;
    pack_wmma_b<<<(total + 255) / 256, 256, 0, stream>>>(W1, W1f, kDin, kDh);
  }
  {
    int total = kDh * kDout;
    pack_wmma_b<<<(total + 255) / 256, 256, 0, stream>>>(W2, W2f, kDh, kDout);
  }
  node_fused<<<kBatch / kBT, kThreads, kDynBytes, stream>>>(
      state, neigh, gamma, beta, b1, b2, aw, ab, W1f, W2f, out);
}
